// Seq2Seq_70944269795409
// MI455X (gfx1250) — matrix operand order repaired and checked by reading, not compiled, not run on hardware
//
#include <hip/hip_runtime.h>
#include <hip/hip_bf16.h>

// ---------------- problem constants ----------------
#define B_    128
#define S_    512   // source length
#define T_    128   // target length (decoder runs T_-1 = 127 steps)
#define E_    128
#define H_    256
#define G4_   1024  // 4*H
#define HD_   512
#define GD_   2048  // 4*HD
#define DENC_ 512
#define KENC_ 384   // E + H
#define XDIM_ 640   // E + DENC
#define KDEC_ 1152  // XDIM + HD
#define VOUT_ 128

typedef __bf16 bf16_t;
typedef __attribute__((ext_vector_type(16))) __bf16 v16bf;
typedef __attribute__((ext_vector_type(8)))  __bf16 v8bf;
typedef __attribute__((ext_vector_type(8)))  float  v8f;

// ---------------- helpers ----------------
__device__ __forceinline__ bf16_t f2bf(float f) {
  union { float f; unsigned u; } v; v.f = f;
  unsigned r = (v.u + 0x7FFFu + ((v.u >> 16) & 1u)) >> 16;
  unsigned short s = (unsigned short)r;
  bf16_t o; __builtin_memcpy(&o, &s, 2);
  return o;
}
__device__ __forceinline__ float sigm(float x) { return 1.0f / (1.0f + expf(-x)); }

// A-operand fragment (16x32 bf16 tile, contiguous k = 16*(lane/16) + s), loaded as two
// aligned 16-byte chunks. Logical A is (M,K) row-major split at column ks
// between A0 and A1 (split points are multiples of 8 -> chunks never straddle).
__device__ __forceinline__ v16bf load_a(const bf16_t* __restrict__ A0, int lda0, int ks,
                                        const bf16_t* __restrict__ A1, int lda1,
                                        int row, int k0, int abase) {
  int k1 = k0 + 2 * abase;   // 16 halves: lanes<16 get K k0..k0+15, lanes>=16 k0+16..31
  int k2 = k1 + 8;
  const bf16_t* p1 = (k1 < ks) ? (A0 + (size_t)row * lda0 + k1)
                               : (A1 + (size_t)row * lda1 + (k1 - ks));
  const bf16_t* p2 = (k2 < ks) ? (A0 + (size_t)row * lda0 + k2)
                               : (A1 + (size_t)row * lda1 + (k2 - ks));
  v8bf lo = *(const v8bf*)p1;
  v8bf hi = *(const v8bf*)p2;
  return __builtin_shufflevector(lo, hi, 0, 1, 2, 3, 4, 5, 6, 7,
                                 8, 9, 10, 11, 12, 13, 14, 15);
}

// Strip GEMM: one wave computes a 16 x (16*NT) output strip. B is pre-packed in
// WMMA register layout: tile (kt,nt) = 32 lanes x 16 contiguous halves, so each
// B fragment is one aligned 32-byte per-lane load.
template <int NT>
__device__ __forceinline__ void wmma_strip(const bf16_t* __restrict__ A0, int lda0, int ks,
                                           const bf16_t* __restrict__ A1, int lda1,
                                           const bf16_t* __restrict__ Bp, int Ntiles,
                                           int K, int m0, int nt0, int lane, v8f acc[NT]) {
  const int row   = m0 + (lane & 15);
  const int abase = (lane >> 4) * 8;
  for (int k0 = 0; k0 < K; k0 += 32) {
    v16bf a = load_a(A0, lda0, ks, A1, lda1, row, k0, abase);
    const bf16_t* bbase =
        Bp + ((((size_t)(k0 >> 5) * Ntiles + nt0) * 32) + lane) * 16;
#pragma unroll
    for (int j = 0; j < NT; ++j) {
      v16bf b = *(const v16bf*)(bbase + (size_t)j * 32 * 16);
      acc[j] = __builtin_amdgcn_wmma_f32_16x16x32_bf16(false, a, false, b,
                                                       (short)0, acc[j], false, false);
    }
  }
}

// ---------------- prep kernels ----------------
// Pack weight into WMMA-B tiled layout.
// trans==1: src is (N,Ksub) row-major (PyTorch-style W: out x in)
// trans==0: src is (Ksub,N) row-major
// koff: k-offset of this sub-block within the full-K packed buffer.
__global__ void k_pack_b(const float* __restrict__ src, bf16_t* __restrict__ dst,
                         int Ksub, int N, int ldsrc, int trans, int koff) {
  int idx = blockIdx.x * blockDim.x + threadIdx.x;  // Ksub*N
  if (idx >= Ksub * N) return;
  int n = idx % N, kl = idx / N;
  float v = trans ? src[(size_t)n * ldsrc + kl] : src[(size_t)kl * ldsrc + n];
  int kg = koff + kl;
  int kt = kg >> 5, kr = kg & 31;
  int lane = ((kr >> 4) << 4) | (n & 15);
  int i = kr & 15;
  int nt = n >> 4;
  dst[((((size_t)kt * (N >> 4) + nt) * 32) + lane) * 16 + i] = f2bf(v);
}

__global__ void k_convert(const float* __restrict__ src, bf16_t* __restrict__ dst, int n) {
  int idx = blockIdx.x * blockDim.x + threadIdx.x;
  if (idx < n) dst[idx] = f2bf(src[idx]);
}

__global__ void k_zero_f(float* p, int n) {
  int idx = blockIdx.x * blockDim.x + threadIdx.x;
  if (idx < n) p[idx] = 0.0f;
}
__global__ void k_zero_bf(bf16_t* p, int n) {
  int idx = blockIdx.x * blockDim.x + threadIdx.x;
  if (idx < n) p[idx] = f2bf(0.0f);
}

// xemb[(t*B + b)*E + e] = enc_emb[src[b][t]][e]   (bf16)
__global__ void k_gather_emb(const int* __restrict__ src, const float* __restrict__ emb,
                             bf16_t* __restrict__ xemb) {
  int idx = blockIdx.x * blockDim.x + threadIdx.x;  // S_*B_*E_
  if (idx >= S_ * B_ * E_) return;
  int e = idx % E_;
  int b = (idx / E_) % B_;
  int t = idx / (B_ * E_);
  xemb[idx] = f2bf(emb[(size_t)src[b * S_ + t] * E_ + e]);
}

// ---------------- encoder ----------------
// gates = [emb_t | h_prev] @ W ; NT=2 => 256 waves/dir (grid (32,2) x 256thr)
__global__ void k_enc_step(int t, const bf16_t* __restrict__ xemb,
                           const bf16_t* __restrict__ hbfF, const bf16_t* __restrict__ hbfB,
                           const bf16_t* __restrict__ WpF, const bf16_t* __restrict__ WpB,
                           float* __restrict__ gatesF, float* __restrict__ gatesB) {
  int dir = blockIdx.y;
  int s = dir ? (S_ - 1 - t) : t;
  const bf16_t* xe  = xemb + (size_t)s * B_ * E_;
  const bf16_t* hbf = dir ? hbfB : hbfF;
  const bf16_t* Wp  = dir ? WpB : WpF;
  float* gates      = dir ? gatesB : gatesF;

  int wid  = blockIdx.x * (blockDim.x >> 5) + (threadIdx.x >> 5);
  int lane = threadIdx.x & 31;
  int mt = wid & 7, st = wid >> 3;           // 8 m-tiles x 32 strips
  int m0 = mt * 16, nt0 = st * 2;
  v8f acc[2] = {};
  wmma_strip<2>(xe, E_, E_, hbf, H_, Wp, G4_ / 16, KENC_, m0, nt0, lane, acc);
  int cn = lane & 15, cm = m0 + (lane >> 4) * 8;
#pragma unroll
  for (int j = 0; j < 2; ++j)
#pragma unroll
    for (int r = 0; r < 8; ++r)
      gates[(size_t)(cm + r) * G4_ + (nt0 + j) * 16 + cn] = acc[j][r];
}

__global__ void k_enc_activate(int t, const float* __restrict__ gatesF,
                               const float* __restrict__ gatesB,
                               const float* __restrict__ biasF, const float* __restrict__ biasB,
                               float* __restrict__ hF, float* __restrict__ cF,
                               float* __restrict__ hB, float* __restrict__ cB,
                               bf16_t* __restrict__ hbfF, bf16_t* __restrict__ hbfB,
                               bf16_t* __restrict__ encOut) {
  int dir = blockIdx.y;
  int idx = blockIdx.x * blockDim.x + threadIdx.x;  // B_*H_
  if (idx >= B_ * H_) return;
  int b = idx / H_, j = idx % H_;
  const float* g    = (dir ? gatesB : gatesF) + (size_t)b * G4_;
  const float* bias = dir ? biasB : biasF;
  float* hv = dir ? hB : hF;
  float* cv = dir ? cB : cF;
  bf16_t* hb = dir ? hbfB : hbfF;

  float ig = sigm(g[j] + bias[j]);
  float fg = sigm(g[H_ + j] + bias[H_ + j]);
  float gg = tanhf(g[2 * H_ + j] + bias[2 * H_ + j]);
  float og = sigm(g[3 * H_ + j] + bias[3 * H_ + j]);
  float c  = fg * cv[idx] + ig * gg;
  float h  = og * tanhf(c);
  cv[idx] = c;
  hv[idx] = h;
  hb[idx] = f2bf(h);
  int s = dir ? (S_ - 1 - t) : t;
  encOut[((size_t)b * S_ + s) * DENC_ + dir * H_ + j] = f2bf(h);
}

// enc_proj (+attn_b folded), stored TRANSPOSED as encPt[b][e][s] so the scores
// kernel reads lane-contiguous in s. NT=4 => 32768 waves.
__global__ void k_enc_proj(const bf16_t* __restrict__ encOut, const bf16_t* __restrict__ WeP,
                           const float* __restrict__ attn_b, bf16_t* __restrict__ encPt) {
  int wid  = blockIdx.x * (blockDim.x >> 5) + (threadIdx.x >> 5);
  int lane = threadIdx.x & 31;
  int st = wid & 7, mt = wid >> 3;           // 8 strips x 4096 m-tiles
  int m0 = mt * 16, nt0 = st * 4;
  v8f acc[4] = {};
  wmma_strip<4>(encOut, DENC_, 1 << 30, encOut, DENC_, WeP, HD_ / 16, DENC_,
                m0, nt0, lane, acc);
  int cn = lane & 15, cmb = m0 + (lane >> 4) * 8;
#pragma unroll
  for (int j = 0; j < 4; ++j) {
    int e = (nt0 + j) * 16 + cn;
    float bb = attn_b[e];
#pragma unroll
    for (int r = 0; r < 8; ++r) {
      int rowM = cmb + r;                    // = b*S + s
      int b = rowM >> 9, s = rowM & (S_ - 1);
      encPt[((size_t)b * HD_ + e) * S_ + s] = f2bf(acc[j][r] + bb);
    }
  }
}

__global__ void k_init_dec(const float* __restrict__ hF, const float* __restrict__ hB,
                           const float* __restrict__ cF, const float* __restrict__ cB,
                           bf16_t* __restrict__ hdecBf, float* __restrict__ cdec) {
  int idx = blockIdx.x * blockDim.x + threadIdx.x;
  if (idx >= B_ * HD_) return;
  int b = idx / HD_, j = idx % HD_;
  float h = (j < H_) ? hF[b * H_ + j] : hB[b * H_ + (j - H_)];
  float c = (j < H_) ? cF[b * H_ + j] : cB[b * H_ + (j - H_)];
  hdecBf[idx] = f2bf(h);
  cdec[idx]   = c;
}

// ---------------- decoder ----------------
// q = h @ Wh   (128 x 512 x 512), NT=2 => 128 waves
__global__ void k_dec_q(const bf16_t* __restrict__ hdecBf, const bf16_t* __restrict__ WhP,
                        float* __restrict__ q) {
  int wid  = blockIdx.x * (blockDim.x >> 5) + (threadIdx.x >> 5);
  int lane = threadIdx.x & 31;
  int mt = wid & 7, st = wid >> 3;           // 8 m-tiles x 16 strips
  int m0 = mt * 16, nt0 = st * 2;
  v8f acc[2] = {};
  wmma_strip<2>(hdecBf, HD_, 1 << 30, hdecBf, HD_, WhP, HD_ / 16, HD_, m0, nt0, lane, acc);
  int cn = lane & 15, cm = m0 + (lane >> 4) * 8;
#pragma unroll
  for (int j = 0; j < 2; ++j)
#pragma unroll
    for (int r = 0; r < 8; ++r)
      q[(size_t)(cm + r) * HD_ + (nt0 + j) * 16 + cn] = acc[j][r];
}

// scores[b][s] = sum_e v[e]*tanh(q[b][e] + encPt[b][e][s]) ; coalesced in s
__global__ void k_scores(const float* __restrict__ q, const bf16_t* __restrict__ encPt,
                         const float* __restrict__ attn_v, float* __restrict__ scores) {
  int idx = blockIdx.x * blockDim.x + threadIdx.x;  // B_*S_
  if (idx >= B_ * S_) return;
  int b = idx >> 9, s = idx & (S_ - 1);
  const bf16_t* ep = encPt + (size_t)b * HD_ * S_ + s;
  const float* qb  = q + (size_t)b * HD_;
  float acc = 0.0f;
  for (int e = 0; e < HD_; ++e)
    acc += attn_v[e] * tanhf(qb[e] + (float)ep[(size_t)e * S_]);
  scores[idx] = acc;
}

__global__ void k_softmax(const float* __restrict__ scores, bf16_t* __restrict__ aw) {
  __shared__ float red[S_];
  int b = blockIdx.x, s = threadIdx.x;
  float v = scores[(size_t)b * S_ + s];
  red[s] = v;
  __syncthreads();
  for (int off = S_ / 2; off > 0; off >>= 1) {
    if (s < off) red[s] = fmaxf(red[s], red[s + off]);
    __syncthreads();
  }
  float mx = red[0];
  __syncthreads();
  float e = expf(v - mx);
  red[s] = e;
  __syncthreads();
  for (int off = S_ / 2; off > 0; off >>= 1) {
    if (s < off) red[s] += red[s + off];
    __syncthreads();
  }
  aw[(size_t)b * S_ + s] = f2bf(e / red[0]);
}

// ctx[b][d] = sum_s aw[b][s]*enc_out[b][s][d]; xbuf = [dec_emb[tok] | ctx] bf16
__global__ void k_ctx_x(int t, const int* __restrict__ trg, const bf16_t* __restrict__ decEmbBf,
                        const bf16_t* __restrict__ aw, const bf16_t* __restrict__ encOut,
                        bf16_t* __restrict__ xbuf) {
  int b = blockIdx.x, d = threadIdx.x;  // 512 threads
  const bf16_t* a  = aw + (size_t)b * S_;
  const bf16_t* eo = encOut + (size_t)b * S_ * DENC_ + d;
  float acc = 0.0f;
  for (int s = 0; s < S_; ++s)
    acc += (float)a[s] * (float)eo[(size_t)s * DENC_];
  xbuf[(size_t)b * XDIM_ + E_ + d] = f2bf(acc);
  if (d < E_) {
    int tok = trg[b * T_ + t];
    xbuf[(size_t)b * XDIM_ + d] = decEmbBf[(size_t)tok * E_ + d];
  }
}

// gates = [x | h] @ Wdec   (128 x 2048 x 1152), NT=4 => 256 waves
__global__ void k_dec_gates(const bf16_t* __restrict__ xbuf, const bf16_t* __restrict__ hdecBf,
                            const bf16_t* __restrict__ WpD, float* __restrict__ gatesD) {
  int wid  = blockIdx.x * (blockDim.x >> 5) + (threadIdx.x >> 5);
  int lane = threadIdx.x & 31;
  int mt = wid & 7, st = wid >> 3;           // 8 m-tiles x 32 strips
  int m0 = mt * 16, nt0 = st * 4;
  v8f acc[4] = {};
  wmma_strip<4>(xbuf, XDIM_, XDIM_, hdecBf, HD_, WpD, GD_ / 16, KDEC_, m0, nt0, lane, acc);
  int cn = lane & 15, cm = m0 + (lane >> 4) * 8;
#pragma unroll
  for (int j = 0; j < 4; ++j)
#pragma unroll
    for (int r = 0; r < 8; ++r)
      gatesD[(size_t)(cm + r) * GD_ + (nt0 + j) * 16 + cn] = acc[j][r];
}

__global__ void k_dec_activate(const float* __restrict__ gatesD, const float* __restrict__ bias,
                               float* __restrict__ cdec, bf16_t* __restrict__ hdecBf) {
  int idx = blockIdx.x * blockDim.x + threadIdx.x;  // B_*HD_
  if (idx >= B_ * HD_) return;
  int b = idx / HD_, j = idx % HD_;
  const float* g = gatesD + (size_t)b * GD_;
  float ig = sigm(g[j] + bias[j]);
  float fg = sigm(g[HD_ + j] + bias[HD_ + j]);
  float gg = tanhf(g[2 * HD_ + j] + bias[2 * HD_ + j]);
  float og = sigm(g[3 * HD_ + j] + bias[3 * HD_ + j]);
  float c = fg * cdec[idx] + ig * gg;
  float h = og * tanhf(c);
  cdec[idx]   = c;
  hdecBf[idx] = f2bf(h);
}

// logits = h @ out_W + out_b -> d_out[:, t+1, :], NT=2 => 32 waves
__global__ void k_logits(int t, const bf16_t* __restrict__ hdecBf, const bf16_t* __restrict__ outWp,
                         const float* __restrict__ out_b, float* __restrict__ out) {
  int wid  = blockIdx.x * (blockDim.x >> 5) + (threadIdx.x >> 5);
  int lane = threadIdx.x & 31;
  int mt = wid & 7, st = wid >> 3;           // 8 m-tiles x 4 strips
  int m0 = mt * 16, nt0 = st * 2;
  v8f acc[2] = {};
  wmma_strip<2>(hdecBf, HD_, 1 << 30, hdecBf, HD_, outWp, VOUT_ / 16, HD_, m0, nt0, lane, acc);
  int cn = lane & 15, cm = m0 + (lane >> 4) * 8;
#pragma unroll
  for (int j = 0; j < 2; ++j) {
    int n = (nt0 + j) * 16 + cn;
    float bb = out_b[n];
#pragma unroll
    for (int r = 0; r < 8; ++r)
      out[((size_t)(cm + r) * T_ + (t + 1)) * VOUT_ + n] = acc[j][r] + bb;
  }
}

__global__ void k_zero_t0(float* __restrict__ out) {
  int idx = blockIdx.x * blockDim.x + threadIdx.x;  // B_*VOUT_
  if (idx >= B_ * VOUT_) return;
  int b = idx / VOUT_, v = idx % VOUT_;
  out[(size_t)b * T_ * VOUT_ + v] = 0.0f;
}

// ---------------- host ----------------
extern "C" void kernel_launch(void* const* d_in, const int* in_sizes, int n_in,
                              void* d_out, int out_size, void* d_ws, size_t ws_size,
                              hipStream_t stream) {
  (void)in_sizes; (void)n_in; (void)out_size; (void)ws_size;

  const int*   src      = (const int*)d_in[0];
  const int*   trg      = (const int*)d_in[1];
  const float* enc_emb  = (const float*)d_in[2];
  const float* enc_WihF = (const float*)d_in[3];
  const float* enc_WhhF = (const float*)d_in[4];
  const float* enc_bF   = (const float*)d_in[5];
  const float* enc_WihB = (const float*)d_in[6];
  const float* enc_WhhB = (const float*)d_in[7];
  const float* enc_bB   = (const float*)d_in[8];
  const float* dec_emb  = (const float*)d_in[9];
  const float* dec_Wih  = (const float*)d_in[10];
  const float* dec_Whh  = (const float*)d_in[11];
  const float* dec_b    = (const float*)d_in[12];
  const float* attn_W   = (const float*)d_in[13];
  const float* attn_b   = (const float*)d_in[14];
  const float* attn_v   = (const float*)d_in[15];
  const float* out_W    = (const float*)d_in[16];
  const float* out_b    = (const float*)d_in[17];
  float* out = (float*)d_out;

  // bump allocator over workspace
  char* p = (char*)d_ws;
  auto alloc = [&](size_t bytes) -> void* {
    void* r = (void*)p;
    p += (bytes + 255) & ~(size_t)255;
    return r;
  };
  bf16_t* WpF      = (bf16_t*)alloc((size_t)KENC_ * G4_ * 2);   // packed enc fwd
  bf16_t* WpB      = (bf16_t*)alloc((size_t)KENC_ * G4_ * 2);   // packed enc bwd
  bf16_t* WpD      = (bf16_t*)alloc((size_t)KDEC_ * GD_ * 2);   // packed dec
  bf16_t* WhP      = (bf16_t*)alloc((size_t)HD_ * HD_ * 2);     // packed attn Wh
  bf16_t* WeP      = (bf16_t*)alloc((size_t)DENC_ * HD_ * 2);   // packed attn We
  bf16_t* outWp    = (bf16_t*)alloc((size_t)HD_ * VOUT_ * 2);   // packed out_W
  bf16_t* decEmbBf = (bf16_t*)alloc((size_t)VOUT_ * E_ * 2);
  bf16_t* xemb     = (bf16_t*)alloc((size_t)S_ * B_ * E_ * 2);
  bf16_t* encOut   = (bf16_t*)alloc((size_t)B_ * S_ * DENC_ * 2);
  bf16_t* encPt    = (bf16_t*)alloc((size_t)B_ * HD_ * S_ * 2); // transposed proj
  bf16_t* hbfF     = (bf16_t*)alloc((size_t)B_ * H_ * 2);
  bf16_t* hbfB     = (bf16_t*)alloc((size_t)B_ * H_ * 2);
  bf16_t* hdecBf   = (bf16_t*)alloc((size_t)B_ * HD_ * 2);
  bf16_t* awBf     = (bf16_t*)alloc((size_t)B_ * S_ * 2);
  bf16_t* xbuf     = (bf16_t*)alloc((size_t)B_ * XDIM_ * 2);
  float*  gatesF   = (float*)alloc((size_t)B_ * G4_ * 4);
  float*  gatesB   = (float*)alloc((size_t)B_ * G4_ * 4);
  float*  hF       = (float*)alloc((size_t)B_ * H_ * 4);
  float*  cF       = (float*)alloc((size_t)B_ * H_ * 4);
  float*  hB       = (float*)alloc((size_t)B_ * H_ * 4);
  float*  cB       = (float*)alloc((size_t)B_ * H_ * 4);
  float*  cdec     = (float*)alloc((size_t)B_ * HD_ * 4);
  float*  qbuf     = (float*)alloc((size_t)B_ * HD_ * 4);
  float*  scores   = (float*)alloc((size_t)B_ * S_ * 4);
  float*  gatesD   = (float*)alloc((size_t)B_ * GD_ * 4);

  const int TB = 256;
  auto nb = [](int n, int t) { return (n + t - 1) / t; };

  // ---- prep: pack weights into WMMA-B layout (bf16) ----
  k_pack_b<<<nb(E_ * G4_, TB), TB, 0, stream>>>(enc_WihF, WpF, E_, G4_, E_, 1, 0);
  k_pack_b<<<nb(H_ * G4_, TB), TB, 0, stream>>>(enc_WhhF, WpF, H_, G4_, H_, 1, E_);
  k_pack_b<<<nb(E_ * G4_, TB), TB, 0, stream>>>(enc_WihB, WpB, E_, G4_, E_, 1, 0);
  k_pack_b<<<nb(H_ * G4_, TB), TB, 0, stream>>>(enc_WhhB, WpB, H_, G4_, H_, 1, E_);
  k_pack_b<<<nb(XDIM_ * GD_, TB), TB, 0, stream>>>(dec_Wih, WpD, XDIM_, GD_, XDIM_, 1, 0);
  k_pack_b<<<nb(HD_ * GD_, TB), TB, 0, stream>>>(dec_Whh, WpD, HD_, GD_, HD_, 1, XDIM_);
  k_pack_b<<<nb(HD_ * HD_, TB), TB, 0, stream>>>(attn_W, WhP, HD_, HD_, HD_, 0, 0);
  k_pack_b<<<nb(DENC_ * HD_, TB), TB, 0, stream>>>(attn_W + (size_t)HD_ * HD_, WeP,
                                                   DENC_, HD_, HD_, 0, 0);
  k_pack_b<<<nb(HD_ * VOUT_, TB), TB, 0, stream>>>(out_W, outWp, HD_, VOUT_, VOUT_, 0, 0);
  k_convert<<<nb(VOUT_ * E_, TB), TB, 0, stream>>>(dec_emb, decEmbBf, VOUT_ * E_);
  k_gather_emb<<<nb(S_ * B_ * E_, TB), TB, 0, stream>>>(src, enc_emb, xemb);

  // ---- zero initial LSTM state ----
  k_zero_f<<<nb(B_ * H_, TB), TB, 0, stream>>>(hF, B_ * H_);
  k_zero_f<<<nb(B_ * H_, TB), TB, 0, stream>>>(cF, B_ * H_);
  k_zero_f<<<nb(B_ * H_, TB), TB, 0, stream>>>(hB, B_ * H_);
  k_zero_f<<<nb(B_ * H_, TB), TB, 0, stream>>>(cB, B_ * H_);
  k_zero_bf<<<nb(B_ * H_, TB), TB, 0, stream>>>(hbfF, B_ * H_);
  k_zero_bf<<<nb(B_ * H_, TB), TB, 0, stream>>>(hbfB, B_ * H_);

  // ---- encoder scan (fwd + bwd fused via blockIdx.y) ----
  for (int t = 0; t < S_; ++t) {
    k_enc_step<<<dim3(32, 2), TB, 0, stream>>>(t, xemb, hbfF, hbfB, WpF, WpB, gatesF, gatesB);
    k_enc_activate<<<dim3(nb(B_ * H_, TB), 2), TB, 0, stream>>>(
        t, gatesF, gatesB, enc_bF, enc_bB, hF, cF, hB, cB, hbfF, hbfB, encOut);
  }

  // ---- enc_proj (one big WMMA GEMM, attn_b folded, transposed store) ----
  k_enc_proj<<<(B_ * S_ / 16) * (HD_ / 16) / 4 / 8, TB, 0, stream>>>(encOut, WeP, attn_b, encPt);

  // ---- decoder init + zero output row 0 ----
  k_init_dec<<<nb(B_ * HD_, TB), TB, 0, stream>>>(hF, hB, cF, cB, hdecBf, cdec);
  k_zero_t0<<<nb(B_ * VOUT_, TB), TB, 0, stream>>>(out);

  // ---- decoder scan ----
  for (int t = 0; t < T_ - 1; ++t) {
    k_dec_q<<<16, TB, 0, stream>>>(hdecBf, WhP, qbuf);
    k_scores<<<nb(B_ * S_, TB), TB, 0, stream>>>(qbuf, encPt, attn_v, scores);
    k_softmax<<<B_, S_, 0, stream>>>(scores, awBf);
    k_ctx_x<<<B_, DENC_, 0, stream>>>(t, trg, decEmbBf, awBf, encOut, xbuf);
    k_dec_gates<<<32, TB, 0, stream>>>(xbuf, hdecBf, WpD, gatesD);
    k_dec_activate<<<nb(B_ * HD_, TB), TB, 0, stream>>>(gatesD, dec_b, cdec, hdecBf);
    k_logits<<<4, TB, 0, stream>>>(t, hdecBf, outWp, out_b, out);
  }
}
